// MultiHeadSelfAttention_9363028705769
// MI455X (gfx1250) — matrix operand order repaired and checked by reading, not compiled, not run on hardware
//
#include <hip/hip_runtime.h>

// ---------------------------------------------------------------------------
// MI455X (gfx1250) causal multi-head attention forward, f16 WMMA pipeline.
//   x[8,1024,1024] @ qkv_w[1024,3072] + b  -> qkv (f16, ws)
//   flash-attn per (b, head, 16-row q tile), causal, online softmax -> ctx (f16, ws)
//   ctx @ out_w[1024,1024] + b             -> out (f32, d_out)
// Matmuls: v_wmma_f32_16x16x32_f16. f16 tile staging: global_load_async_to_lds.
// ---------------------------------------------------------------------------

typedef _Float16 h8  __attribute__((ext_vector_type(8)));
typedef _Float16 h16 __attribute__((ext_vector_type(16)));
typedef float    f8  __attribute__((ext_vector_type(8)));

union Frag {
    h16 v;
    h8  half8s[2];
    _Float16 e[16];
};

#define WMMA_F16(a, b, c) \
    __builtin_amdgcn_wmma_f32_16x16x32_f16(false, (a), false, (b), (short)0, (c), false, false)

// Wave-local LDS fence (in-order DS pipe within a wave; no block barrier —
// attention waves have divergent causal trip counts).
__device__ __forceinline__ void lds_fence() {
    asm volatile("s_wait_dscnt 0x0" ::: "memory");
}
__device__ __forceinline__ void async_fence() {
    asm volatile("s_wait_asynccnt 0x0" ::: "memory");
}

// Async DMA: copy 128 contiguous bytes / lane, global -> LDS (ASYNCcnt).
// IOFFSET applies to both the LDS and global addresses (ISA 15.18.3).
__device__ __forceinline__ void async_copy_128B(unsigned lds_addr, const void* gaddr) {
    asm volatile(
        "global_load_async_to_lds_b128 %0, %1, off\n\t"
        "global_load_async_to_lds_b128 %0, %1, off offset:16\n\t"
        "global_load_async_to_lds_b128 %0, %1, off offset:32\n\t"
        "global_load_async_to_lds_b128 %0, %1, off offset:48\n\t"
        "global_load_async_to_lds_b128 %0, %1, off offset:64\n\t"
        "global_load_async_to_lds_b128 %0, %1, off offset:80\n\t"
        "global_load_async_to_lds_b128 %0, %1, off offset:96\n\t"
        "global_load_async_to_lds_b128 %0, %1, off offset:112"
        :: "v"(lds_addr), "v"(gaddr) : "memory");
}
// Async DMA: copy 32 contiguous bytes / lane, global -> LDS.
__device__ __forceinline__ void async_copy_32B(unsigned lds_addr, const void* gaddr) {
    asm volatile(
        "global_load_async_to_lds_b128 %0, %1, off\n\t"
        "global_load_async_to_lds_b128 %0, %1, off offset:16"
        :: "v"(lds_addr), "v"(gaddr) : "memory");
}
// Low 32 bits of a generic shared pointer are the LDS byte offset.
__device__ __forceinline__ unsigned lds_off(const void* p) {
    return (unsigned)(size_t)p;
}

// ---------------------------------------------------------------------------
// GEMM: C[M,N] = A[M,K] * B[K,N] + bias[N]
//   A: f32 (converted in staging) or f16 (async-DMA'd). B: f32. C: f16 or f32.
//   Block = 256 threads (8 waves), tile 128x128, BK=32.
//   Wave w -> (wm = w>>2, wn = w&3): 64x32 output = 4x2 WMMA tiles (8 WMMA/step).
// ---------------------------------------------------------------------------
template <bool A_F16, bool C_F16>
__global__ __launch_bounds__(256) void gemm_bias_kernel(
    const void* __restrict__ Av, const float* __restrict__ B,
    const float* __restrict__ bias, void* __restrict__ Cv,
    int M, int N, int K)
{
    constexpr int BM = 128, BN = 128, BK = 32;
    constexpr int AS = BK + 8;   // padded LDS row strides (halves)
    constexpr int BS = BK + 8;
    __shared__ _Float16 As[BM * AS];
    __shared__ _Float16 Bt[BN * BS];   // B tile stored transposed: Bt[n][k]

    const int tid   = threadIdx.x;
    const int wave  = tid >> 5;
    const int lane  = tid & 31;
    const int lrow  = lane & 15;   // m / n index inside a 16-wide fragment
    const int lhalf = lane >> 4;   // 0: lanes 0-15, 1: lanes 16-31
    const int wm    = wave >> 2;   // 0..1  (64 rows each)
    const int wn    = wave & 3;    // 0..3  (32 cols each)
    const int mbase = blockIdx.y * BM;
    const int nbase = blockIdx.x * BN;

    const float*    Af = (const float*)Av;
    const _Float16* Ah = (const _Float16*)Av;

    const int ar = tid >> 1;             // A staging: row 0..127
    const int ac = (tid & 1) * 16;       // A staging: col 0 or 16
    const int bk = tid >> 3;             // B staging: k row 0..31
    const int bn = (tid & 7) * 16;       // B staging: n col 0..112

    f8 acc[4][2] = {};

    for (int k0 = 0; k0 < K; k0 += BK) {
        // ---- stage A tile (128 x 32) ----
        if (A_F16) {
            async_copy_32B(lds_off(&As[ar * AS + ac]),
                           Ah + (size_t)(mbase + ar) * K + k0 + ac);
        } else {
            const float* src = Af + (size_t)(mbase + ar) * K + k0 + ac;
            _Float16* dst = &As[ar * AS + ac];
            #pragma unroll
            for (int i = 0; i < 16; ++i) dst[i] = (_Float16)src[i];
        }
        // ---- stage B tile transposed: B[k0+bk][nbase+bn..+15] -> Bt[n][k] ----
        {
            const float* src = B + (size_t)(k0 + bk) * N + nbase + bn;
            #pragma unroll
            for (int i = 0; i < 16; ++i) Bt[(bn + i) * BS + bk] = (_Float16)src[i];
        }
        // prefetch next k-tile source rows
        if (k0 + BK < K) {
            if (A_F16) __builtin_prefetch(Ah + (size_t)(mbase + ar) * K + k0 + BK + ac, 0, 1);
            else       __builtin_prefetch(Af + (size_t)(mbase + ar) * K + k0 + BK + ac, 0, 1);
            __builtin_prefetch(B + (size_t)(k0 + BK + bk) * N + nbase + bn, 0, 1);
        }
        if (A_F16) async_fence();   // own async DMA done before the barrier
        __syncthreads();

        // ---- fragments ----
        // A (16x32 f16): m = lrow; halves 0-7 -> K = lhalf*8+i, 8-15 -> 16+lhalf*8+i.
        Frag a[4];
        #pragma unroll
        for (int mt = 0; mt < 4; ++mt) {
            const _Float16* p = &As[(wm * 64 + mt * 16 + lrow) * AS];
            a[mt].half8s[0] = *(const h8*)(p + lhalf * 8);
            a[mt].half8s[1] = *(const h8*)(p + 16 + lhalf * 8);
        }
        // B (32x16 f16): n = lrow; halves 0-7 -> K = lhalf*8+i, 8-15 -> 16+lhalf*8+i.
        Frag b[2];
        #pragma unroll
        for (int nt = 0; nt < 2; ++nt) {
            const _Float16* p = &Bt[(wn * 32 + nt * 16 + lrow) * BS];
            b[nt].half8s[0] = *(const h8*)(p + lhalf * 8);
            b[nt].half8s[1] = *(const h8*)(p + 16 + lhalf * 8);
        }
        #pragma unroll
        for (int mt = 0; mt < 4; ++mt)
            #pragma unroll
            for (int nt = 0; nt < 2; ++nt)
                acc[mt][nt] = WMMA_F16(a[mt].v, b[nt].v, acc[mt][nt]);

        __syncthreads();
    }

    // ---- epilogue: bias + store (C/D layout: col = lrow, row = r + 8*lhalf) ----
    #pragma unroll
    for (int nt = 0; nt < 2; ++nt) {
        const int col = nbase + wn * 32 + nt * 16 + lrow;
        const float bv = bias[col];
        #pragma unroll
        for (int mt = 0; mt < 4; ++mt) {
            #pragma unroll
            for (int r = 0; r < 8; ++r) {
                const int row = mbase + wm * 64 + mt * 16 + r + lhalf * 8;
                const float v = acc[mt][nt][r] + bv;
                if (C_F16) ((_Float16*)Cv)[(size_t)row * N + col] = (_Float16)v;
                else       ((float*)Cv)[(size_t)row * N + col] = v;
            }
        }
    }
}

// ---------------------------------------------------------------------------
// Flash attention (causal), one wave per (b, head, 16-row q tile).
//   qkv: [8, 1024, 3072] f16, row = [q(16 heads x 64) | k(...) | v(...)]
//   ctx: [8, 1024, 1024] f16
// Per K-tile of 32: S = Q K^T (4 WMMA), online softmax, O += P V (4 WMMA).
// V tile staged to LDS via async DMA, overlapped with the S WMMAs.
// ---------------------------------------------------------------------------
__global__ __launch_bounds__(256) void attn_kernel(
    const _Float16* __restrict__ qkv, _Float16* __restrict__ ctx)
{
    constexpr int SEQn = 1024, D = 1024, DS = 64, RS = 3 * D;
    const float scale = 0.125f;   // 1/sqrt(64)

    __shared__ _Float16 Vs[8][32 * 64];   // per-wave V tile (32 k-rows x 64 d)
    __shared__ _Float16 Ps[8][16 * 32];   // per-wave P re-layout buffer

    const int tid   = threadIdx.x;
    const int wave  = tid >> 5;
    const int lane  = tid & 31;
    const int lrow  = lane & 15;
    const int lhalf = lane >> 4;

    const int gw    = blockIdx.x * 8 + wave;   // 8192 waves total
    const int qt    = gw & 63;                 // q tile (SEQ/16 = 64)
    const int hh    = (gw >> 6) & 15;          // head
    const int bb    = gw >> 10;                // batch
    const int qbase = qt * 16;

    const _Float16* qp = qkv + (size_t)bb * SEQn * RS + hh * DS;
    const _Float16* kp = qp + D;
    const _Float16* vp = qp + 2 * D;

    // Q as two A-fragments (d chunks 0-31 / 32-63); m = lrow = q row in tile.
    Frag aq[2];
    #pragma unroll
    for (int c = 0; c < 2; ++c) {
        const _Float16* row = qp + (size_t)(qbase + lrow) * RS + c * 32;
        aq[c].half8s[0] = *(const h8*)(row + lhalf * 8);
        aq[c].half8s[1] = *(const h8*)(row + 16 + lhalf * 8);
    }

    const unsigned vs_lds = lds_off(&Vs[wave][lane * 64]);

    f8 o[4] = {};
    float mrun[8], lsum[8];
    #pragma unroll
    for (int r = 0; r < 8; ++r) { mrun[r] = -1e30f; lsum[r] = 0.0f; }

    const int nkt = qt / 2 + 1;   // causal: K tiles with kbase <= qbase
    for (int kt = 0; kt < nkt; ++kt) {
        const int kb = kt * 32;

        // ---- async-DMA V[kb..kb+31][0..63] into LDS (one 128B row per lane) ----
        async_copy_128B(vs_lds, vp + (size_t)(kb + lane) * RS);

        // ---- S = Q K^T : two 16-col tiles, each 2 WMMAs over d (overlaps DMA) ----
        f8 s[2];
        #pragma unroll
        for (int nt = 0; nt < 2; ++nt) {
            const _Float16* krow = kp + (size_t)(kb + nt * 16 + lrow) * RS;
            Frag bk0, bk1;   // B fragment: n = lrow = k-row; K-dim = d, table order
            bk0.half8s[0] = *(const h8*)(krow + lhalf * 8);
            bk0.half8s[1] = *(const h8*)(krow + 16 + lhalf * 8);
            bk1.half8s[0] = *(const h8*)(krow + 32 + lhalf * 8);
            bk1.half8s[1] = *(const h8*)(krow + 48 + lhalf * 8);
            f8 z = {};
            z = WMMA_F16(aq[0].v, bk0.v, z);
            z = WMMA_F16(aq[1].v, bk1.v, z);
            s[nt] = z;
        }

        // ---- online softmax (row stats live in the lanes that own the row) ----
        float p0[8], p1[8], alpha[8];
        #pragma unroll
        for (int r = 0; r < 8; ++r) {
            const int row = qbase + r + lhalf * 8;
            float s0 = s[0][r] * scale;
            float s1 = s[1][r] * scale;
            if (kb + lrow > row)      s0 = -1e30f;   // causal mask
            if (kb + 16 + lrow > row) s1 = -1e30f;
            float rm = fmaxf(s0, s1);
            #pragma unroll
            for (int off = 1; off < 16; off <<= 1)
                rm = fmaxf(rm, __shfl_xor(rm, off, 16));
            const float mnew = fmaxf(mrun[r], rm);
            const float a    = __expf(mrun[r] - mnew);
            const float e0   = __expf(s0 - mnew);
            const float e1   = __expf(s1 - mnew);
            float rs = e0 + e1;
            #pragma unroll
            for (int off = 1; off < 16; off <<= 1)
                rs += __shfl_xor(rs, off, 16);
            mrun[r]  = mnew;
            lsum[r]  = lsum[r] * a + rs;
            alpha[r] = a;
            p0[r] = e0;
            p1[r] = e1;
        }
        #pragma unroll
        for (int dt = 0; dt < 4; ++dt)
            #pragma unroll
            for (int r = 0; r < 8; ++r) o[dt][r] *= alpha[r];

        // ---- P (C/D layout) -> LDS row-major -> A fragment ----
        #pragma unroll
        for (int r = 0; r < 8; ++r) {
            Ps[wave][(r + lhalf * 8) * 32 + lrow]      = (_Float16)p0[r];
            Ps[wave][(r + lhalf * 8) * 32 + 16 + lrow] = (_Float16)p1[r];
        }
        async_fence();   // V DMA landed in LDS
        lds_fence();     // P stores visible wave-wide
        Frag ap;
        ap.half8s[0] = *(const h8*)(&Ps[wave][lrow * 32 + lhalf * 16]);
        ap.half8s[1] = *(const h8*)(&Ps[wave][lrow * 32 + lhalf * 16 + 8]);

        // ---- O += P V : 4 d-tiles of 16 ----
        #pragma unroll
        for (int dt = 0; dt < 4; ++dt) {
            Frag bv;   // n = lrow = d col; K = k index, contiguous-16 per lane
            #pragma unroll
            for (int i = 0; i < 16; ++i)
                bv.e[i] = Vs[wave][(lhalf * 16 + i) * 64 + dt * 16 + lrow];
            o[dt] = WMMA_F16(ap.v, bv.v, o[dt]);
        }
        lds_fence();   // drain DS reads before next iteration overwrites Vs/Ps
    }

    // ---- normalize, store context (f16) ----
    #pragma unroll
    for (int dt = 0; dt < 4; ++dt) {
        #pragma unroll
        for (int r = 0; r < 8; ++r) {
            const int row = qbase + r + lhalf * 8;
            const float v = o[dt][r] * (1.0f / lsum[r]);
            ctx[(size_t)(bb * SEQn + row) * D + hh * DS + dt * 16 + lrow] = (_Float16)v;
        }
    }
}

// ---------------------------------------------------------------------------
extern "C" void kernel_launch(void* const* d_in, const int* in_sizes, int n_in,
                              void* d_out, int out_size, void* d_ws, size_t ws_size,
                              hipStream_t stream) {
    (void)in_sizes; (void)n_in; (void)out_size; (void)ws_size;
    const float* x     = (const float*)d_in[0];   // [8,1024,1024]
    const float* qkv_w = (const float*)d_in[1];   // [1024,3072]
    const float* qkv_b = (const float*)d_in[2];   // [3072]
    const float* out_w = (const float*)d_in[3];   // [1024,1024]
    const float* out_b = (const float*)d_in[4];   // [1024]

    const int M = 8 * 1024, Dm = 1024, N3 = 3072;
    _Float16* qkv_h = (_Float16*)d_ws;                    // 48 MB
    _Float16* ctx_h = qkv_h + (size_t)M * N3;             // +16 MB

    dim3 blk(256);
    // QKV projection: [8192,1024] x [1024,3072] -> f16
    gemm_bias_kernel<false, true><<<dim3(N3 / 128, M / 128), blk, 0, stream>>>(
        x, qkv_w, qkv_b, qkv_h, M, N3, Dm);
    // Causal flash attention: 8192 waves / 8 per block
    attn_kernel<<<dim3(1024), blk, 0, stream>>>(qkv_h, ctx_h);
    // Output projection: [8192,1024] x [1024,1024] -> f32
    gemm_bias_kernel<true, false><<<dim3(Dm / 128, M / 128), blk, 0, stream>>>(
        ctx_h, out_w, out_b, d_out, M, Dm, Dm);
}
